// DeepseekV2ExpertsFix_19095424598381
// MI455X (gfx1250) — matrix operand order repaired and checked by reading, not compiled, not run on hardware
//
#include <hip/hip_runtime.h>
#include <hip/hip_bf16.h>

// ---------------- problem constants ----------------
constexpr int kE = 16;    // experts
constexpr int kH = 2048;  // hidden
constexpr int kI = 1408;  // intermediate
constexpr int kT = 2048;  // tokens
constexpr int kK = 6;     // top-k

constexpr int TILE_M    = 32;            // tokens per block tile (2 x 16-row WMMA tiles)
constexpr int NWAVES    = 8;             // 256 threads / wave32
constexpr int HS_STRIDE = kH + 8;        // bf16 elems, keeps 16B alignment
constexpr int ACT_STRIDE= kI + 8;        // bf16 elems

typedef __attribute__((ext_vector_type(16))) __bf16 v16bf;
typedef __attribute__((ext_vector_type(8)))  float  v8f;

union FragBF {
  v16bf v;
  unsigned int u[8];
  uint4 q[2];
};

// fp32 -> bf16 pair using native conversions (v_cvt_pk_bf16_f32 on gfx1250)
__device__ __forceinline__ unsigned int pack_bf2(float a, float b) {
  union { __bf16 h[2]; unsigned int u; } r;
  r.h[0] = (__bf16)a;
  r.h[1] = (__bf16)b;
  return r.u;
}

__device__ __forceinline__ unsigned short f2bf(float a) {
  union { __bf16 h; unsigned short u; } r;
  r.h = (__bf16)a;
  return r.u;
}

// fast silu: g * rcp(1+exp(-g)) -> v_exp_f32 + v_rcp_f32 (bf16-accurate)
__device__ __forceinline__ float fast_silu(float g) {
  return g * __builtin_amdgcn_rcpf(1.0f + __expf(-g));
}

// ---- B-fragment loaders: 16 contiguous K elements per lane ----
__device__ __forceinline__ void load_bfrag(const unsigned short* __restrict__ p, FragBF& b) {
  b.q[0] = *(const uint4*)p;       // 8 bf16
  b.q[1] = *(const uint4*)(p + 8); // 8 bf16
}
__device__ __forceinline__ void load_bfrag(const float* __restrict__ p, FragBF& b) {
  const float4* p4 = (const float4*)p;
  float4 f0 = p4[0], f1 = p4[1], f2 = p4[2], f3 = p4[3];
  b.u[0] = pack_bf2(f0.x, f0.y); b.u[1] = pack_bf2(f0.z, f0.w);
  b.u[2] = pack_bf2(f1.x, f1.y); b.u[3] = pack_bf2(f1.z, f1.w);
  b.u[4] = pack_bf2(f2.x, f2.y); b.u[5] = pack_bf2(f2.z, f2.w);
  b.u[6] = pack_bf2(f3.x, f3.y); b.u[7] = pack_bf2(f3.z, f3.w);
}

// ---------------- kernel 1: zero output + routing counters ----------------
__global__ void zero_init_kernel(float* __restrict__ out, int n, int* __restrict__ counts) {
  int i = blockIdx.x * blockDim.x + threadIdx.x;
  if (i < kE) counts[i] = 0;
  if (i < n)  out[i] = 0.0f;
}

// ---------------- kernel 2: build per-expert compacted token lists --------
__global__ void build_routing_kernel(const int* __restrict__ idx,
                                     const float* __restrict__ w,
                                     int* __restrict__ counts,
                                     int* __restrict__ toks,
                                     float* __restrict__ wgts) {
  int t = blockIdx.x * blockDim.x + threadIdx.x;
  if (t >= kT) return;
  #pragma unroll
  for (int k = 0; k < kK; ++k) {
    int e = idx[t * kK + k];
    bool dup = false;
    for (int j = 0; j < k; ++j)
      if (idx[t * kK + j] == e) dup = true;   // merged into earlier slot
    if (dup) continue;
    float wsum = 0.0f;
    for (int j = k; j < kK; ++j)
      if (idx[t * kK + j] == e) wsum += w[t * kK + j];
    int slot = atomicAdd(&counts[e], 1);
    toks[e * kT + slot] = t;
    wgts[e * kT + slot] = wsum;
  }
}

// ---------------- kernel 3: bulk fp32 -> bf16 conversion ------------------
__global__ void cvt_bf16_kernel(const float* __restrict__ src,
                                unsigned short* __restrict__ dst,
                                long n8) {
  long i = (long)blockIdx.x * blockDim.x + threadIdx.x;
  if (i >= n8) return;
  long o = i * 8;
  const float4* p = (const float4*)(src + o);
  float4 a = p[0], b = p[1];
  uint4 q;
  q.x = pack_bf2(a.x, a.y); q.y = pack_bf2(a.z, a.w);
  q.z = pack_bf2(b.x, b.y); q.w = pack_bf2(b.z, b.w);
  *(uint4*)(dst + o) = q;
}

// ---------------- kernel 4: fused gathered MoE expert ---------------------
// grid = (ceil(T/32) token tiles, E experts), block = 256 threads (8 wave32)
// WT = unsigned short (preconverted bf16 weights) or float (convert on the fly)
template <typename WT>
__global__ __launch_bounds__(256, 1)
void moe_fused_kernel(const unsigned short* __restrict__ hsb,   // [T,H] bf16
                      const WT* __restrict__ gu,                // [E,2I,H]
                      const WT* __restrict__ dn,                // [E,H,I]
                      const int*   __restrict__ counts,
                      const int*   __restrict__ toks,           // [E,T]
                      const float* __restrict__ wgts,           // [E,T]
                      float* __restrict__ out) {                // [T,H] f32
  __shared__ unsigned short hstile[TILE_M * HS_STRIDE];   // 32 x 2048 bf16 (+pad) ~128.5KB
  __shared__ unsigned short act[TILE_M * ACT_STRIDE];     // 32 x 1408 bf16 (+pad) ~88.5KB
  __shared__ int   stok[TILE_M];
  __shared__ float swgt[TILE_M];

  const int e   = blockIdx.y;
  const int mt  = blockIdx.x;
  const int cnt = counts[e];
  if (mt * TILE_M >= cnt) return;           // uniform across the block

  const int tid  = threadIdx.x;
  const int wave = tid >> 5;
  const int lane = tid & 31;
  const int col  = lane & 15;               // N column (and A row) for this lane
  const int hi   = lane >> 4;               // high half-wave

  if (tid < TILE_M) {
    int s = mt * TILE_M + tid;
    if (s < cnt) { stok[tid] = toks[e * kT + s]; swgt[tid] = wgts[e * kT + s]; }
    else         { stok[tid] = toks[e * kT];     swgt[tid] = 0.0f; }  // padded row, weight 0
  }
  __syncthreads();

  // stage 32 gathered hidden rows into LDS — reused by all N-tiles of phase 1
  for (int idx = tid; idx < TILE_M * (kH / 8); idx += blockDim.x) {
    int row = idx >> 8;                 // kH/8 == 256 chunks per row
    int c8  = (idx & 255) * 8;
    uint4 v = *(const uint4*)(hsb + (size_t)stok[row] * kH + c8);
    *(uint4*)(hstile + row * HS_STRIDE + c8) = v;
  }
  __syncthreads();

  const WT* guE = gu + (size_t)e * (2 * kI) * kH;
  const WT* dnE = dn + (size_t)e * kH * kI;

  // ---- phase 1: act[32,I] = silu(x Wg^T) * (x Wu^T); waves stride over N-tiles ----
  for (int nt = wave; nt < kI / 16; nt += NWAVES) {
    const int ng = nt * 16 + col;         // gate row of gate_up_proj
    const int nu = kI + nt * 16 + col;    // up   row of gate_up_proj
    v8f cg0 = {0.f,0.f,0.f,0.f,0.f,0.f,0.f,0.f};
    v8f cg1 = {0.f,0.f,0.f,0.f,0.f,0.f,0.f,0.f};
    v8f cu0 = {0.f,0.f,0.f,0.f,0.f,0.f,0.f,0.f};
    v8f cu1 = {0.f,0.f,0.f,0.f,0.f,0.f,0.f,0.f};
    const WT* pgr = guE + (size_t)ng * kH;
    const WT* pur = guE + (size_t)nu * kH;
    #pragma unroll 2
    for (int k0 = 0; k0 < kH; k0 += 32) {
      // A: lanes0-15 -> K {0..15}; lanes16-31 -> K {16..31}
      FragBF a0, a1;
      const unsigned short* pa0 = hstile + col * HS_STRIDE + k0;
      const unsigned short* pa1 = hstile + (16 + col) * HS_STRIDE + k0;
      a0.q[0] = *(const uint4*)(pa0 + hi * 16);
      a0.q[1] = *(const uint4*)(pa0 + hi * 16 + 8);
      a1.q[0] = *(const uint4*)(pa1 + hi * 16);
      a1.q[1] = *(const uint4*)(pa1 + hi * 16 + 8);
      FragBF bg, bu;  // B: lane col = N, 16 contiguous K per lane
      load_bfrag(pgr + k0 + hi * 16, bg);
      load_bfrag(pur + k0 + hi * 16, bu);
      cg0 = __builtin_amdgcn_wmma_f32_16x16x32_bf16(false, a0.v, false, bg.v,
                                                    (short)0, cg0, false, false);
      cg1 = __builtin_amdgcn_wmma_f32_16x16x32_bf16(false, a1.v, false, bg.v,
                                                    (short)0, cg1, false, false);
      cu0 = __builtin_amdgcn_wmma_f32_16x16x32_bf16(false, a0.v, false, bu.v,
                                                    (short)0, cu0, false, false);
      cu1 = __builtin_amdgcn_wmma_f32_16x16x32_bf16(false, a1.v, false, bu.v,
                                                    (short)0, cu1, false, false);
    }
    #pragma unroll
    for (int r = 0; r < 8; ++r) {
      int row = r + hi * 8;
      act[row * ACT_STRIDE + nt * 16 + col]        = f2bf(fast_silu(cg0[r]) * cu0[r]);
      act[(16 + row) * ACT_STRIDE + nt * 16 + col] = f2bf(fast_silu(cg1[r]) * cu1[r]);
    }
  }
  __syncthreads();

  // ---- phase 2: out[32,H] += w * (act @ down^T); waves stride over H-tiles ----
  for (int ht = wave; ht < kH / 16; ht += NWAVES) {
    const int hcol = ht * 16 + col;               // row of down_proj
    v8f c0 = {0.f,0.f,0.f,0.f,0.f,0.f,0.f,0.f};
    v8f c1 = {0.f,0.f,0.f,0.f,0.f,0.f,0.f,0.f};
    const WT* pdr = dnE + (size_t)hcol * kI;
    #pragma unroll 2
    for (int k0 = 0; k0 < kI; k0 += 32) {
      FragBF a0, a1;
      const unsigned short* pa0 = act + col * ACT_STRIDE + k0;
      const unsigned short* pa1 = act + (16 + col) * ACT_STRIDE + k0;
      a0.q[0] = *(const uint4*)(pa0 + hi * 16);
      a0.q[1] = *(const uint4*)(pa0 + hi * 16 + 8);
      a1.q[0] = *(const uint4*)(pa1 + hi * 16);
      a1.q[1] = *(const uint4*)(pa1 + hi * 16 + 8);
      FragBF b;
      load_bfrag(pdr + k0 + hi * 16, b);
      c0 = __builtin_amdgcn_wmma_f32_16x16x32_bf16(false, a0.v, false, b.v,
                                                   (short)0, c0, false, false);
      c1 = __builtin_amdgcn_wmma_f32_16x16x32_bf16(false, a1.v, false, b.v,
                                                   (short)0, c1, false, false);
    }
    #pragma unroll
    for (int r = 0; r < 8; ++r) {
      int row = r + hi * 8;
      float v0 = c0[r] * swgt[row];
      float v1 = c1[r] * swgt[16 + row];
      unsafeAtomicAdd(out + (size_t)stok[row] * kH + hcol, v0);
      unsafeAtomicAdd(out + (size_t)stok[16 + row] * kH + hcol, v1);
    }
  }
}

// ---------------- host-side launch ----------------------------------------
extern "C" void kernel_launch(void* const* d_in, const int* in_sizes, int n_in,
                              void* d_out, int out_size, void* d_ws, size_t ws_size,
                              hipStream_t stream) {
  const float* hs  = (const float*)d_in[0];   // [T,H]
  const int*   idx = (const int*)d_in[1];     // [T,K]
  const float* tw  = (const float*)d_in[2];   // [T,K]
  const float* gu  = (const float*)d_in[3];   // [E,2I,H]
  const float* dnp = (const float*)d_in[4];   // [E,H,I]
  float* out = (float*)d_out;

  // workspace layout
  constexpr size_t OFF_COUNTS = 0;                                  // 64 ints
  constexpr size_t OFF_TOKS   = 256;                                // E*T ints
  constexpr size_t OFF_WGTS   = OFF_TOKS + sizeof(int) * kE * kT;   // E*T floats
  constexpr size_t OFF_HSB    = OFF_WGTS + sizeof(float) * kE * kT; // T*H bf16
  constexpr size_t OFF_GUB    = OFF_HSB + (size_t)2 * kT * kH;      // E*2I*H bf16
  constexpr size_t OFF_DNB    = OFF_GUB + (size_t)2 * kE * 2 * kI * kH;
  constexpr size_t WS_FULL    = OFF_DNB + (size_t)2 * kE * kH * kI; // ~273 MB

  char* ws = (char*)d_ws;
  int*   counts = (int*)(ws + OFF_COUNTS);
  int*   toks   = (int*)(ws + OFF_TOKS);
  float* wgts   = (float*)(ws + OFF_WGTS);
  unsigned short* hsb = (unsigned short*)(ws + OFF_HSB);
  unsigned short* gub = (unsigned short*)(ws + OFF_GUB);
  unsigned short* dnb = (unsigned short*)(ws + OFF_DNB);

  const int nOut = kT * kH;
  zero_init_kernel<<<(nOut + 255) / 256, 256, 0, stream>>>(out, nOut, counts);
  build_routing_kernel<<<(kT + 127) / 128, 128, 0, stream>>>(idx, tw, counts, toks, wgts);

  long nHs8 = (long)kT * kH / 8;
  cvt_bf16_kernel<<<(unsigned)((nHs8 + 255) / 256), 256, 0, stream>>>(hs, hsb, nHs8);

  dim3 grid((kT + TILE_M - 1) / TILE_M, kE);   // 64 token tiles x 16 experts

  if (ws_size >= WS_FULL) {
    // preconvert weights to bf16 once: halves weight bytes and removes per-tile VALU
    long nGu8 = (long)kE * 2 * kI * kH / 8;
    long nDn8 = (long)kE * kH * kI / 8;
    cvt_bf16_kernel<<<(unsigned)((nGu8 + 255) / 256), 256, 0, stream>>>(gu, gub, nGu8);
    cvt_bf16_kernel<<<(unsigned)((nDn8 + 255) / 256), 256, 0, stream>>>(dnp, dnb, nDn8);
    moe_fused_kernel<unsigned short><<<grid, 256, 0, stream>>>(
        hsb, gub, dnb, counts, toks, wgts, out);
  } else {
    // fallback: convert weights in-register per tile
    moe_fused_kernel<float><<<grid, 256, 0, stream>>>(
        hsb, gu, dnp, counts, toks, wgts, out);
  }
}
